// PeTwoPhaseLSTM_45586782880039
// MI455X (gfx1250) — compile-verified, matrix operand order checked by reading, not run on hardware
//
#include <hip/hip_runtime.h>
#include <hip/hip_bf16.h>
#include <math.h>

// ---------------- model dims ----------------
#define SEQ   80
#define BAT   128
#define HID   512
#define EMB   300
#define EMBP  320           // E padded to multiple of 32 for WMMA K loop
#define G4H   2048          // 4*H gate width
#define STS   79            // st-LSTM steps (S-1)

typedef _Float16 h16v __attribute__((ext_vector_type(16)));
typedef float    f8v  __attribute__((ext_vector_type(8)));

#define GF_ACC  1
#define GF_RELU 2

// Load one 16x32 f16 WMMA fragment (A-row or W-row layout, ISA 7.12.2).
// base = row pointer for this lane (M or N = tile + (lane&15)); hi = lane>>4.
__device__ __forceinline__ h16v load_frag(const _Float16* __restrict__ base,
                                          int k0, int hi)
{
    h16v f;
#pragma unroll
    for (int jj = 0; jj < 8; ++jj) {
        const int kk = k0 + ((jj >> 2) << 4) + (hi << 3) + ((jj & 3) << 1);
        f[2 * jj]     = base[kk];
        f[2 * jj + 1] = base[kk + 1];
    }
    return f;
}

// ---------------- WMMA GEMM: C[M,N] (+)= A[M,K] * W[N,K]^T ----------------
// one wave (32 lanes) per 32x32 output tile (2x2 WMMA register tiling);
// K multiple of 32; M,N multiple of 32.
__global__ __launch_bounds__(32)
void wmma_gemm_kernel(const _Float16* __restrict__ A, int lda,
                      const _Float16* __restrict__ W, int ldw,
                      float* __restrict__ C, int ldc,
                      _Float16* __restrict__ C16, int ldc16,
                      const float* __restrict__ bias,
                      int K, int flags)
{
    const int lane = threadIdx.x & 31;
    const int l15  = lane & 15;
    const int hi   = lane >> 4;           // 0: K 0-7/16-23 half, 1: K 8-15/24-31 half
    const int tm   = blockIdx.y << 5;
    const int tn   = blockIdx.x << 5;

    const _Float16* Ar0 = A + (size_t)(tm + l15) * lda;
    const _Float16* Ar1 = A + (size_t)(tm + 16 + l15) * lda;
    const _Float16* Wr0 = W + (size_t)(tn + l15) * ldw;
    const _Float16* Wr1 = W + (size_t)(tn + 16 + l15) * ldw;

    f8v acc00 = {}, acc01 = {}, acc10 = {}, acc11 = {};
    if (flags & GF_ACC) {
#pragma unroll
        for (int j = 0; j < 8; ++j) {
            const int r0 = tm + hi * 8 + j;
            const int r1 = r0 + 16;
            acc00[j] = C[(size_t)r0 * ldc + (tn + l15)];
            acc01[j] = C[(size_t)r0 * ldc + (tn + 16 + l15)];
            acc10[j] = C[(size_t)r1 * ldc + (tn + l15)];
            acc11[j] = C[(size_t)r1 * ldc + (tn + 16 + l15)];
        }
    }

    for (int k0 = 0; k0 < K; k0 += 32) {
        const h16v a0 = load_frag(Ar0, k0, hi);
        const h16v a1 = load_frag(Ar1, k0, hi);
        const h16v b0 = load_frag(Wr0, k0, hi);
        const h16v b1 = load_frag(Wr1, k0, hi);
        if (k0 + 32 < K) {                 // cover latency of next K block
            __builtin_prefetch(Ar0 + k0 + 32, 0, 0);   // global_prefetch_b8
            __builtin_prefetch(Ar1 + k0 + 32, 0, 0);
            __builtin_prefetch(Wr0 + k0 + 32, 0, 0);
            __builtin_prefetch(Wr1 + k0 + 32, 0, 0);
        }
        acc00 = __builtin_amdgcn_wmma_f32_16x16x32_f16(
                    false, a0, false, b0, (short)0, acc00, false, false);
        acc01 = __builtin_amdgcn_wmma_f32_16x16x32_f16(
                    false, a0, false, b1, (short)0, acc01, false, false);
        acc10 = __builtin_amdgcn_wmma_f32_16x16x32_f16(
                    false, a1, false, b0, (short)0, acc10, false, false);
        acc11 = __builtin_amdgcn_wmma_f32_16x16x32_f16(
                    false, a1, false, b1, (short)0, acc11, false, false);
    }

#pragma unroll
    for (int j = 0; j < 8; ++j) {
        const int r0 = tm + hi * 8 + j;
        const int r1 = r0 + 16;
        const int c0 = tn + l15;
        const int c1 = c0 + 16;
        float v00 = acc00[j], v01 = acc01[j], v10 = acc10[j], v11 = acc11[j];
        if (bias) {
            const float bb0 = bias[c0], bb1 = bias[c1];
            v00 += bb0; v01 += bb1; v10 += bb0; v11 += bb1;
        }
        if (flags & GF_RELU) {
            v00 = v00 > 0.f ? v00 : 0.f;  v01 = v01 > 0.f ? v01 : 0.f;
            v10 = v10 > 0.f ? v10 : 0.f;  v11 = v11 > 0.f ? v11 : 0.f;
        }
        if (C) {
            C[(size_t)r0 * ldc + c0] = v00;  C[(size_t)r0 * ldc + c1] = v01;
            C[(size_t)r1 * ldc + c0] = v10;  C[(size_t)r1 * ldc + c1] = v11;
        }
        if (C16) {
            C16[(size_t)r0 * ldc16 + c0] = (_Float16)v00;
            C16[(size_t)r0 * ldc16 + c1] = (_Float16)v01;
            C16[(size_t)r1 * ldc16 + c0] = (_Float16)v10;
            C16[(size_t)r1 * ldc16 + c1] = (_Float16)v11;
        }
    }
}

// ---------------- utility kernels ----------------
__global__ void zero_f32_kernel(float* p, long n) {
    long i = (long)blockIdx.x * blockDim.x + threadIdx.x;
    if (i < n) p[i] = 0.f;
}
__global__ void zero_f16_kernel(_Float16* p, long n) {
    long i = (long)blockIdx.x * blockDim.x + threadIdx.x;
    if (i < n) p[i] = (_Float16)0.f;
}

// dst[n, 0:ldd] = f16(src[n, koff : koff+ksrc]) zero-padded to ldd
__global__ void cvt_f16_pad_kernel(_Float16* __restrict__ dst, int ldd,
                                   const float* __restrict__ src, int lds,
                                   int koff, int ksrc, long total)
{
    long idx = (long)blockIdx.x * blockDim.x + threadIdx.x;
    if (idx >= total) return;
    int  k = (int)(idx % ldd);
    long n = idx / ldd;
    dst[idx] = (k < ksrc) ? (_Float16)src[n * (long)lds + koff + k] : (_Float16)0.f;
}

// ex16[(s*B+b), e] = emb[x_src[s*B+b], e], padded 300 -> 320
__global__ void embed_kernel(const int* __restrict__ x,
                             const float* __restrict__ emb,
                             _Float16* __restrict__ ex16)
{
    long idx = (long)blockIdx.x * blockDim.x + threadIdx.x;
    if (idx >= (long)SEQ * BAT * EMBP) return;
    int  e = (int)(idx % EMBP);
    long r = idx / EMBP;
    int tok = x[r];
    ex16[idx] = (e < EMB) ? (_Float16)emb[(size_t)tok * EMB + e] : (_Float16)0.f;
}

// sinusoidal PE table (80 x 512); both even/odd columns use exponent 2*d/H
__global__ void pe_kernel(float* __restrict__ pe)
{
    int idx = blockIdx.x * blockDim.x + threadIdx.x;
    if (idx >= SEQ * HID) return;
    int p = idx / HID, d = idx % HID;
    float ang = (float)p * __powf(10000.f, -2.f * (float)d / (float)HID);
    pe[idx] = (d & 1) ? __cosf(ang) : __sinf(ang);
}

// LSTM cell pointwise: gates = gx(f16) + gh + bih + bhh, split i|f|g|o
__global__ void lstm_cell_kernel(const _Float16* __restrict__ gx,
                                 const float* __restrict__ gh,
                                 const float* __restrict__ bih,
                                 const float* __restrict__ bhh,
                                 float* __restrict__ h, float* __restrict__ c,
                                 _Float16* __restrict__ h16,
                                 float* __restrict__ hpe, _Float16* __restrict__ hpe16,
                                 const float* __restrict__ pe, float scale,
                                 float* __restrict__ hsum)
{
    int idx = blockIdx.x * blockDim.x + threadIdx.x;
    if (idx >= BAT * HID) return;
    int b = idx >> 9, j = idx & (HID - 1);
    size_t base = (size_t)b * G4H;
    float gi = (float)gx[base + j]            + gh[base + j]            + bih[j]            + bhh[j];
    float gf = (float)gx[base + HID + j]      + gh[base + HID + j]      + bih[HID + j]      + bhh[HID + j];
    float gg = (float)gx[base + 2 * HID + j]  + gh[base + 2 * HID + j]  + bih[2 * HID + j]  + bhh[2 * HID + j];
    float go = (float)gx[base + 3 * HID + j]  + gh[base + 3 * HID + j]  + bih[3 * HID + j]  + bhh[3 * HID + j];
    float si = 1.f / (1.f + __expf(-gi));
    float sf = 1.f / (1.f + __expf(-gf));
    float so = 1.f / (1.f + __expf(-go));
    float cn = sf * c[idx] + si * tanhf(gg);
    float hn = so * tanhf(cn);
    c[idx] = cn;
    h[idx] = hn;
    h16[idx] = (_Float16)hn;
    if (hpe) {
        float pv = hn * scale + pe[j];
        hpe[idx] = pv;
        if (hpe16) hpe16[idx] = (_Float16)pv;
    }
    if (hsum) hsum[idx] += hn;
}

// scores[b,s] = (q[b] . k[s,b]) / sqrt(H); one wave per (s,b)
__global__ __launch_bounds__(32)
void attn_scores_kernel(const float* __restrict__ q, const float* __restrict__ k,
                        float* __restrict__ sc)
{
    int s = blockIdx.x, b = blockIdx.y, lane = threadIdx.x;
    const float* qb = q + (size_t)b * HID;
    const float* kr = k + ((size_t)s * BAT + b) * HID;
    float acc = 0.f;
    for (int j = lane; j < HID; j += 32) acc += qb[j] * kr[j];
    for (int o = 16; o > 0; o >>= 1) acc += __shfl_down(acc, o, 32);
    if (lane == 0) sc[b * SEQ + s] = acc * 0.044194173824159216f;   // 1/sqrt(512)
}

__global__ void softmax_rows_kernel(float* __restrict__ sc)
{
    int b = blockIdx.x * blockDim.x + threadIdx.x;
    if (b >= BAT) return;
    float* r = sc + (size_t)b * SEQ;
    float m = r[0];
    for (int s = 1; s < SEQ; ++s) m = fmaxf(m, r[s]);
    float sum = 0.f;
    for (int s = 0; s < SEQ; ++s) { float e = __expf(r[s] - m); r[s] = e; sum += e; }
    float inv = 1.f / sum;
    for (int s = 0; s < SEQ; ++s) r[s] *= inv;
}

// a_pre[b,j] = sum_s w[b,s] * v[s,b,j]   (f16 out for next GEMM)
__global__ void attn_wsum_kernel(const float* __restrict__ w,
                                 const float* __restrict__ v,
                                 _Float16* __restrict__ ap16)
{
    int idx = blockIdx.x * blockDim.x + threadIdx.x;
    if (idx >= BAT * HID) return;
    int b = idx >> 9, j = idx & (HID - 1);
    float acc = 0.f;
    for (int s = 0; s < SEQ; ++s)
        acc += w[b * SEQ + s] * v[((size_t)s * BAT + b) * HID + j];
    ap16[idx] = (_Float16)acc;
}

// final heads: one wave per batch element; out[0:384]=y_stance, out[384:768]=y_sub
__global__ __launch_bounds__(32)
void head_kernel(const float* __restrict__ st_sum, const float* __restrict__ hpe,
                 const float* __restrict__ w1, const float* __restrict__ b1,
                 const float* __restrict__ w2, const float* __restrict__ b2,
                 float* __restrict__ out)
{
    int b = blockIdx.x, lane = threadIdx.x;
    float s0 = 0, s1 = 0, s2 = 0, t0 = 0, t1 = 0, t2 = 0;
    for (int j = lane; j < HID; j += 32) {
        float mh = st_sum[(size_t)b * HID + j] * (1.f / (float)STS);
        s0 += mh * w1[j]; s1 += mh * w1[HID + j]; s2 += mh * w1[2 * HID + j];
        float acc = 0.f;
        for (int s = 0; s < SEQ; ++s) acc += hpe[((size_t)s * BAT + b) * HID + j];
        acc *= (1.f / (float)SEQ);
        t0 += acc * w2[j]; t1 += acc * w2[HID + j]; t2 += acc * w2[2 * HID + j];
    }
    for (int o = 16; o > 0; o >>= 1) {
        s0 += __shfl_down(s0, o, 32); s1 += __shfl_down(s1, o, 32); s2 += __shfl_down(s2, o, 32);
        t0 += __shfl_down(t0, o, 32); t1 += __shfl_down(t1, o, 32); t2 += __shfl_down(t2, o, 32);
    }
    if (lane == 0) {
        s0 += b1[0]; s1 += b1[1]; s2 += b1[2];
        float m  = fmaxf(s0, fmaxf(s1, s2));
        float ls = m + __logf(__expf(s0 - m) + __expf(s1 - m) + __expf(s2 - m));
        out[b * 3 + 0] = s0 - ls; out[b * 3 + 1] = s1 - ls; out[b * 3 + 2] = s2 - ls;
        t0 += b2[0]; t1 += b2[1]; t2 += b2[2];
        float m2  = fmaxf(t0, fmaxf(t1, t2));
        float ls2 = m2 + __logf(__expf(t0 - m2) + __expf(t1 - m2) + __expf(t2 - m2));
        out[BAT * 3 + b * 3 + 0] = t0 - ls2;
        out[BAT * 3 + b * 3 + 1] = t1 - ls2;
        out[BAT * 3 + b * 3 + 2] = t2 - ls2;
    }
}

// ---------------- host orchestration ----------------
extern "C" void kernel_launch(void* const* d_in, const int* in_sizes, int n_in,
                              void* d_out, int out_size, void* d_ws, size_t ws_size,
                              hipStream_t stream)
{
    (void)in_sizes; (void)n_in; (void)out_size; (void)ws_size;
    const int*   x_src    = (const int*)  d_in[0];
    const float* emb      = (const float*)d_in[1];
    const float* sub_Wih  = (const float*)d_in[2];
    const float* sub_Whh  = (const float*)d_in[3];
    const float* sub_bih  = (const float*)d_in[4];
    const float* sub_bhh  = (const float*)d_in[5];
    const float* ff_W1    = (const float*)d_in[6];
    const float* ff_b1    = (const float*)d_in[7];
    const float* ff_W2    = (const float*)d_in[8];
    const float* ff_b2    = (const float*)d_in[9];
    const float* attn_Win = (const float*)d_in[10];
    const float* attn_bin = (const float*)d_in[11];
    const float* attn_Wout= (const float*)d_in[12];
    const float* attn_bout= (const float*)d_in[13];
    const float* st_Wih   = (const float*)d_in[14];
    const float* st_Whh   = (const float*)d_in[15];
    const float* st_bih   = (const float*)d_in[16];
    const float* st_bhh   = (const float*)d_in[17];
    const float* lin1_W   = (const float*)d_in[18];
    const float* lin1_b   = (const float*)d_in[19];
    const float* lin2_W   = (const float*)d_in[20];
    const float* lin2_b   = (const float*)d_in[21];
    float* out = (float*)d_out;

    // --- carve scratch ---
    char* ws = (char*)d_ws;
    size_t off = 0;
    auto alloc = [&](size_t bytes) -> void* {
        void* p = ws + off;
        off += (bytes + 255) & ~(size_t)255;
        return p;
    };
    const long MB = (long)SEQ * BAT;        // 10240
    const long MB1 = (long)STS * BAT;       // 10112
    _Float16* ex16      = (_Float16*)alloc((size_t)MB * EMBP * 2);
    float*    peb       = (float*)   alloc((size_t)SEQ * HID * 4);
    _Float16* gx_sub16  = (_Float16*)alloc((size_t)MB  * G4H * 2);
    _Float16* gx_st16   = (_Float16*)alloc((size_t)MB1 * G4H * 2);
    _Float16* subWih16  = (_Float16*)alloc((size_t)G4H * EMBP * 2);
    _Float16* subWhh16  = (_Float16*)alloc((size_t)G4H * HID * 2);
    _Float16* ffW1_16   = (_Float16*)alloc((size_t)2 * HID * HID * 2);
    _Float16* ffW2_16   = (_Float16*)alloc((size_t)HID * 2 * HID * 2);
    _Float16* attnWin16 = (_Float16*)alloc((size_t)3 * HID * HID * 2);
    _Float16* attnWout16= (_Float16*)alloc((size_t)HID * HID * 2);
    _Float16* stWihX16  = (_Float16*)alloc((size_t)G4H * EMBP * 2);
    _Float16* stWihA16  = (_Float16*)alloc((size_t)G4H * HID * 2);
    _Float16* stWhh16   = (_Float16*)alloc((size_t)G4H * HID * 2);
    float*    hbuf      = (float*)   alloc((size_t)BAT * HID * 4);
    float*    cbuf      = (float*)   alloc((size_t)BAT * HID * 4);
    _Float16* h16       = (_Float16*)alloc((size_t)BAT * HID * 2);
    float*    gh        = (float*)   alloc((size_t)BAT * G4H * 4);
    float*    hpe       = (float*)   alloc((size_t)MB * HID * 4);
    _Float16* hpe16     = (_Float16*)alloc((size_t)MB * HID * 2);
    _Float16* ffa16     = (_Float16*)alloc((size_t)MB * 2 * HID * 2);
    _Float16* kv16      = (_Float16*)alloc((size_t)MB * HID * 2);
    float*    kbuf      = (float*)   alloc((size_t)MB * HID * 4);
    float*    vbuf      = (float*)   alloc((size_t)MB * HID * 4);
    float*    qbuf      = (float*)   alloc((size_t)BAT * HID * 4);
    float*    scores    = (float*)   alloc((size_t)BAT * SEQ * 4);
    _Float16* apre16    = (_Float16*)alloc((size_t)BAT * HID * 2);
    _Float16* a16       = (_Float16*)alloc((size_t)BAT * HID * 2);
    float*    st_sum    = (float*)   alloc((size_t)BAT * HID * 4);

    auto gemm = [&](const _Float16* A, int lda, const _Float16* W, int ldw,
                    float* C, int ldc, _Float16* C16, int ldc16,
                    const float* bias, int M, int N, int K, int flags) {
        dim3 g(N / 32, M / 32);
        wmma_gemm_kernel<<<g, 32, 0, stream>>>(A, lda, W, ldw, C, ldc, C16, ldc16,
                                               bias, K, flags);
    };
    auto cvt = [&](_Float16* dst, int ldd, const float* src, int lds,
                   int koff, int ksrc, long rows) {
        long total = rows * ldd;
        cvt_f16_pad_kernel<<<(total + 255) / 256, 256, 0, stream>>>(dst, ldd, src, lds,
                                                                    koff, ksrc, total);
    };

    // --- init state + tables ---
    zero_f32_kernel<<<(BAT * HID + 255) / 256, 256, 0, stream>>>(hbuf,  BAT * HID);
    zero_f32_kernel<<<(BAT * HID + 255) / 256, 256, 0, stream>>>(cbuf,  BAT * HID);
    zero_f32_kernel<<<(BAT * HID + 255) / 256, 256, 0, stream>>>(st_sum, BAT * HID);
    zero_f16_kernel<<<(BAT * HID + 255) / 256, 256, 0, stream>>>(h16,   BAT * HID);
    pe_kernel<<<(SEQ * HID + 255) / 256, 256, 0, stream>>>(peb);
    embed_kernel<<<(MB * EMBP + 255) / 256, 256, 0, stream>>>(x_src, emb, ex16);

    // --- convert weights to f16 ---
    cvt(subWih16, EMBP, sub_Wih, EMB, 0, EMB, G4H);
    cvt(subWhh16, HID, sub_Whh, HID, 0, HID, G4H);
    cvt(ffW1_16, HID, ff_W1, HID, 0, HID, 2 * HID);
    cvt(ffW2_16, 2 * HID, ff_W2, 2 * HID, 0, 2 * HID, HID);
    cvt(attnWin16, HID, attn_Win, HID, 0, HID, 3 * HID);
    cvt(attnWout16, HID, attn_Wout, HID, 0, HID, HID);
    cvt(stWihX16, EMBP, st_Wih, EMB + HID, 0, EMB, G4H);
    cvt(stWihA16, HID, st_Wih, EMB + HID, EMB, HID, G4H);
    cvt(stWhh16, HID, st_Whh, HID, 0, HID, G4H);

    // --- hoisted input-side GEMMs for both LSTMs ---
    gemm(ex16, EMBP, subWih16, EMBP, nullptr, 0, gx_sub16, G4H,
         nullptr, (int)MB,  G4H, EMBP, 0);
    gemm(ex16, EMBP, stWihX16, EMBP, nullptr, 0, gx_st16,  G4H,
         nullptr, (int)MB1, G4H, EMBP, 0);

    const float scaleH = sqrtf((float)HID);
    const _Float16* Wq16 = attnWin16;
    const _Float16* Wk16 = attnWin16 + (size_t)HID * HID;
    const _Float16* Wv16 = attnWin16 + (size_t)2 * HID * HID;
    const float* bq = attn_bin;
    const float* bk = attn_bin + HID;
    const float* bv = attn_bin + 2 * HID;

    // --- sub LSTM scan ---
    for (int t = 0; t < SEQ; ++t) {
        gemm(h16, HID, subWhh16, HID, gh, G4H, nullptr, 0,
             nullptr, BAT, G4H, HID, 0);
        lstm_cell_kernel<<<(BAT * HID + 255) / 256, 256, 0, stream>>>(
            gx_sub16 + (size_t)t * BAT * G4H, gh, sub_bih, sub_bhh,
            hbuf, cbuf, h16,
            hpe + (size_t)t * BAT * HID, hpe16 + (size_t)t * BAT * HID,
            peb + (size_t)t * HID, scaleH, nullptr);
    }

    // --- FF chain + K/V projections (big WMMA GEMMs) ---
    gemm(hpe16, HID, ffW1_16, HID, nullptr, 0, ffa16, 2 * HID,
         ff_b1, (int)MB, 2 * HID, HID, GF_RELU);
    gemm(ffa16, 2 * HID, ffW2_16, 2 * HID, nullptr, 0, kv16, HID,
         ff_b2, (int)MB, HID, 2 * HID, GF_RELU);
    gemm(kv16, HID, Wk16, HID, kbuf, HID, nullptr, 0,
         bk, (int)MB, HID, HID, 0);
    gemm(kv16, HID, Wv16, HID, vbuf, HID, nullptr, 0,
         bv, (int)MB, HID, HID, 0);

    // --- st LSTM scan with attention (h,c continue from hT,cT) ---
    for (int t = 0; t < STS; ++t) {
        gemm(h16, HID, Wq16, HID, qbuf, HID, nullptr, 0,
             bq, BAT, HID, HID, 0);
        {
            dim3 g(SEQ, BAT);
            attn_scores_kernel<<<g, 32, 0, stream>>>(qbuf, kbuf, scores);
        }
        softmax_rows_kernel<<<1, 128, 0, stream>>>(scores);
        attn_wsum_kernel<<<(BAT * HID + 255) / 256, 256, 0, stream>>>(scores, vbuf, apre16);
        gemm(apre16, HID, attnWout16, HID, nullptr, 0, a16, HID,
             attn_bout, BAT, HID, HID, 0);
        gemm(a16, HID, stWihA16, HID, gh, G4H, nullptr, 0,
             nullptr, BAT, G4H, HID, 0);
        gemm(h16, HID, stWhh16, HID, gh, G4H, nullptr, 0,
             nullptr, BAT, G4H, HID, GF_ACC);
        lstm_cell_kernel<<<(BAT * HID + 255) / 256, 256, 0, stream>>>(
            gx_st16 + (size_t)t * BAT * G4H, gh, st_bih, st_bhh,
            hbuf, cbuf, h16,
            nullptr, nullptr, nullptr, 0.f, st_sum);
    }

    // --- heads ---
    head_kernel<<<BAT, 32, 0, stream>>>(st_sum, hpe, lin1_W, lin1_b, lin2_W, lin2_b, out);
}
